// AttentionModule_16707422781666
// MI455X (gfx1250) — matrix operand order repaired and checked by reading, not compiled, not run on hardware
//
#include <hip/hip_runtime.h>
#include <hip/hip_bf16.h>
#include <math.h>

typedef __bf16 bf16;
typedef bf16  bf16x8 __attribute__((ext_vector_type(8)));
typedef bf16  v16bf  __attribute__((ext_vector_type(16)));
typedef float v8f    __attribute__((ext_vector_type(8)));
typedef float f32x4  __attribute__((ext_vector_type(4)));
typedef unsigned int u32x4 __attribute__((ext_vector_type(4)));
typedef int   i32x8 __attribute__((ext_vector_type(8)));
typedef int   i32x4 __attribute__((ext_vector_type(4)));

#define NHEADS 16
#define SEQ    2048
#define DMODEL 1024
#define BATCH  4

__device__ __forceinline__ v8f wmma_bf16(v16bf a, v16bf b, v8f c) {
  return __builtin_amdgcn_wmma_f32_16x16x32_bf16(false, a, false, b, (short)0, c, false, false);
}

// Load a 16-element A/B fragment slice as two contiguous 8x bf16 (b128) loads.
__device__ __forceinline__ v16bf load_ab(const bf16* p0, const bf16* p1) {
  bf16x8 lo = *(const bf16x8*)p0;
  bf16x8 hi = *(const bf16x8*)p1;
  return __builtin_shufflevector(lo, hi, 0,1,2,3,4,5,6,7,8,9,10,11,12,13,14,15);
}

// ---- Tensor Data Mover: DMA a [rows x 64B] 2D tile (8-byte units) from global to LDS.
// D# per CDNA5 ISA: Group0 = {count, lds_addr, global_addr, type=2},
// Group1 = {data_size=8B, tensor dims, tile_dim0=8 units, tile_dim1=rows, dim0 stride}.
__device__ __forceinline__ void tdm_load_2d(const bf16* gsrc, void* ldst,
                                            int rows, int stride_units) {
  unsigned long long ga = (unsigned long long)(__SIZE_TYPE__)gsrc;
  unsigned int la = (unsigned int)(__SIZE_TYPE__)ldst;   // low 32 bits = LDS offset
  u32x4 g0;
  g0[0] = 1u;                                            // count = 1 valid descriptor
  g0[1] = la;                                            // lds_addr (bytes)
  g0[2] = (unsigned int)ga;                              // global_addr[31:0]
  g0[3] = (unsigned int)(ga >> 32) | 0x80000000u;        // global_addr[56:32] | type=2<<30
  i32x8 g1;
  g1[0] = 0x00030000;            // workgroup_mask=0, data_size=3 (8 bytes)
  g1[1] = 0;                     // atomic barrier off; tensor_dim0 low16 (1<<20 -> 0)
  g1[2] = 16;                    // tensor_dim0 hi (1<<20>>16); tensor_dim1 low16 = 0
  g1[3] = 16 | (8 << 16);        // tensor_dim1 hi (1<<20>>16); tile_dim0 = 8 units (64B)
  g1[4] = rows & 0xffff;         // tile_dim1 = rows; tile_dim2 = 0
  g1[5] = stride_units;          // tensor_dim0_stride[31:0] (8B units)
  g1[6] = 0;                     // stride0 hi; tensor_dim1_stride low = 0
  g1[7] = 0;
  i32x4 z4; z4[0] = 0; z4[1] = 0; z4[2] = 0; z4[3] = 0;
  i32x8 z8;
#pragma unroll
  for (int i = 0; i < 8; ++i) z8[i] = 0;
  __builtin_amdgcn_tensor_load_to_lds(g0, g1, z4, z4, z8, 0);
}

// ---------------------------------------------------------------- casts
__global__ void cast_f32_bf16(const float* __restrict__ src, bf16* __restrict__ dst, int n) {
  int i = blockIdx.x * blockDim.x + threadIdx.x;
  int stride = gridDim.x * blockDim.x;
  for (; i < n; i += stride) dst[i] = (bf16)src[i];
}

// ---------------------------------------------------------------- bf16 GEMM: C = A[M,K] * W[N,K]^T
// Block = 8 waves sharing one 64-wide N tile; W K-slab (64x32 bf16 = 4KB) staged to LDS
// by the Tensor Data Mover, double-buffered, TENSORcnt-synchronized.
// mode 0: bf16 out, per-head layout  out[((b*H+h)*S + s)*64 + d]
// mode 1: f32  out, row-major [M,N]
// mode 2: bf16 out, per-head transposed out[((b*H+h)*64 + d)*S + s]
__global__ void __launch_bounds__(256)
gemm_bf16_wmma(const bf16* __restrict__ A, const bf16* __restrict__ W,
               void* __restrict__ out, int M, int N, int K, int S, int mode) {
  __shared__ bf16 wbuf[2][64 * 32];                      // 2 x 4KB double buffer

  const int lane = threadIdx.x & 31;
  const int g    = lane >> 4;
  const int ln   = lane & 15;
  const int wid  = threadIdx.x >> 5;
  const int ntiles = N >> 6;
  const int rowb = (blockIdx.x / ntiles) * 128;
  const int col0 = (blockIdx.x % ntiles) * 64;
  const int row0 = rowb + wid * 16;
  const int stride_units = K >> 2;                       // K halves = K*2 B = K/4 x 8B

  v8f acc[4] = {v8f{}, v8f{}, v8f{}, v8f{}};
  const bf16* arow = A + (size_t)(row0 + ln) * K;

  // prologue: stage first W slab
  if (wid == 0) {
    tdm_load_2d(W + (size_t)col0 * K, &wbuf[0][0], 64, stride_units);
    __builtin_amdgcn_s_wait_tensorcnt(0);
  }
  __syncthreads();

  int cur = 0;
  for (int k0 = 0; k0 < K; k0 += 32) {
    const int nxt = cur ^ 1;
    const bool more = (k0 + 32) < K;
    if (wid == 0 && more)
      tdm_load_2d(W + (size_t)col0 * K + k0 + 32, &wbuf[nxt][0], 64, stride_units);

    __builtin_prefetch(arow + k0 + 256, 0, 3);
    // A fragment: lane row = ln; element i -> k = (i<8?i:i+8) + 8*g + k0
    v16bf af = load_ab(arow + k0 + g * 8, arow + k0 + 16 + g * 8);
#pragma unroll
    for (int t = 0; t < 4; ++t) {
      // B fragment from LDS: lane col = ln; element i -> k = (i<8?i:i+8) + 8*g
      const bf16* brow = &wbuf[cur][(t * 16 + ln) * 32 + g * 8];
      v16bf bfr = load_ab(brow, brow + 16);
      acc[t] = wmma_bf16(af, bfr, acc[t]);
    }

    if (wid == 0 && more) __builtin_amdgcn_s_wait_tensorcnt(0);
    __syncthreads();
    cur = nxt;
  }

  // C layout: VGPR r, lanes 0-15 -> row r, lanes 16-31 -> row r+8; col = ln
  if (mode == 1) {
    float* o = (float*)out;
#pragma unroll
    for (int t = 0; t < 4; ++t)
#pragma unroll
      for (int r = 0; r < 8; ++r) {
        int row = row0 + r + 8 * g;
        o[(size_t)row * N + col0 + t * 16 + ln] = acc[t][r];
      }
  } else {
    bf16* o = (bf16*)out;
    const int h = col0 >> 6;
#pragma unroll
    for (int t = 0; t < 4; ++t)
#pragma unroll
      for (int r = 0; r < 8; ++r) {
        int row = row0 + r + 8 * g;          // row = b*S + s
        int b = row / S, s = row - b * S;
        int d = t * 16 + ln;
        size_t idx = (mode == 0)
                   ? ((size_t)(b * NHEADS + h) * S + s) * 64 + d
                   : ((size_t)(b * NHEADS + h) * 64 + d) * (size_t)S + s;
        o[idx] = (bf16)acc[t][r];
      }
  }
}

// ---------------------------------------------------------------- relative-position bias table [16][4096]
__global__ void build_bias_table(const float* __restrict__ rel_emb, float* __restrict__ table) {
  int i = blockIdx.x * blockDim.x + threadIdx.x;
  if (i >= NHEADS * 4096) return;
  int h = i >> 12, idx = i & 4095;
  int d = idx - 2047;                 // relative position k - q
  int rb = (d > 0) ? 16 : 0;
  int rp = (d < 0) ? -d : d;
  int bucket;
  if (rp < 8) {
    bucket = rb + rp;
  } else {
    float l = __logf((float)rp * 0.125f) * (8.0f / 2.7725887222397811f); // log(16)
    int lg = 8 + (int)l;
    bucket = rb + (lg < 15 ? lg : 15);
  }
  table[i] = rel_emb[bucket * NHEADS + h];
}

// ---------------------------------------------------------------- stream position_bias out (pure bandwidth)
__global__ void __launch_bounds__(256)
write_bias(const float* __restrict__ table, float* __restrict__ out) {
  size_t i4 = ((size_t)blockIdx.x * blockDim.x + threadIdx.x) * 4;
  int h = (int)(i4 >> 22);
  int q = (int)((i4 >> 11) & 2047);
  int k = (int)(i4 & 2047);
  const float* t = table + h * 4096 + (k - q + 2047);
  f32x4 v;
  v[0] = t[0]; v[1] = t[1]; v[2] = t[2]; v[3] = t[3];
  __builtin_nontemporal_store(v, (f32x4*)(out + i4));    // don't pollute L2 (K/V live there)
}

// ---------------------------------------------------------------- flash attention, one wave = 16 q rows
__global__ void __launch_bounds__(256)
attn_flash(const bf16* __restrict__ Qb, const bf16* __restrict__ Kb,
           const bf16* __restrict__ Vt, const float* __restrict__ table,
           bf16* __restrict__ ctx, int S) {
  __shared__ float tbl[4096];
  __shared__ bf16  pbuf[8][16 * 32];

  const int bh = blockIdx.y;
  const int h = bh & (NHEADS - 1), b = bh >> 4;
  for (int i = threadIdx.x; i < 4096; i += blockDim.x) tbl[i] = table[h * 4096 + i];
  __syncthreads();

  const int wid  = threadIdx.x >> 5;
  const int lane = threadIdx.x & 31;
  const int g = lane >> 4, ln = lane & 15;
  const int q0 = (blockIdx.x * 8 + wid) * 16;

  // Q as A fragments (dh = 64 -> two K=32 fragments), row = ln
  const bf16* qrow = Qb + ((size_t)bh * S + q0 + ln) * 64;
  v16bf qa0 = load_ab(qrow +  0 + g * 8, qrow + 16 + g * 8);
  v16bf qa1 = load_ab(qrow + 32 + g * 8, qrow + 48 + g * 8);

  v8f acc[4] = {v8f{}, v8f{}, v8f{}, v8f{}};
  float Mr[8], Lr[8];
#pragma unroll
  for (int r = 0; r < 8; ++r) { Mr[r] = -1.0e30f; Lr[r] = 0.0f; }

  const int bias_base = ln - q0 - 8 * g + 2047;

  for (int kc = 0; kc < S; kc += 32) {
    // ---- scores: S[16 x 32] = Q (16x64) * K^T (64x32) via 4 WMMAs
    const bf16* k0p = Kb + ((size_t)bh * S + kc + ln) * 64;
    const bf16* k1p = k0p + 16 * 64;
    v8f s0 = v8f{}, s1 = v8f{};
    {
      v16bf bk;
      bk = load_ab(k0p +  0 + g * 8, k0p + 16 + g * 8); s0 = wmma_bf16(qa0, bk, s0);
      bk = load_ab(k0p + 32 + g * 8, k0p + 48 + g * 8); s0 = wmma_bf16(qa1, bk, s0);
      bk = load_ab(k1p +  0 + g * 8, k1p + 16 + g * 8); s1 = wmma_bf16(qa0, bk, s1);
      bk = load_ab(k1p + 32 + g * 8, k1p + 48 + g * 8); s1 = wmma_bf16(qa1, bk, s1);
    }

    // ---- bias add + online softmax (rows live across 16-lane halves)
    float p0[8], p1[8];
#pragma unroll
    for (int r = 0; r < 8; ++r) {
      p0[r] = s0[r] + tbl[bias_base + kc - r];
      p1[r] = s1[r] + tbl[bias_base + kc + 16 - r];
    }
#pragma unroll
    for (int r = 0; r < 8; ++r) {
      float m = fmaxf(p0[r], p1[r]);
      m = fmaxf(m, __shfl_xor(m, 1, 32));
      m = fmaxf(m, __shfl_xor(m, 2, 32));
      m = fmaxf(m, __shfl_xor(m, 4, 32));
      m = fmaxf(m, __shfl_xor(m, 8, 32));
      float nm = fmaxf(Mr[r], m);
      float al = __expf(Mr[r] - nm);
      float e0 = __expf(p0[r] - nm);
      float e1 = __expf(p1[r] - nm);
      float rs = e0 + e1;
      rs += __shfl_xor(rs, 1, 32);
      rs += __shfl_xor(rs, 2, 32);
      rs += __shfl_xor(rs, 4, 32);
      rs += __shfl_xor(rs, 8, 32);
      Lr[r] = Lr[r] * al + rs;
      Mr[r] = nm;
      p0[r] = e0; p1[r] = e1;
      acc[0][r] *= al; acc[1][r] *= al; acc[2][r] *= al; acc[3][r] *= al;
    }

    // ---- bounce P (C layout) -> LDS -> A-fragment layout
    bf16* pw = &pbuf[wid][0];
#pragma unroll
    for (int r = 0; r < 8; ++r) {
      pw[(r + 8 * g) * 32 + ln]      = (bf16)p0[r];
      pw[(r + 8 * g) * 32 + 16 + ln] = (bf16)p1[r];
    }
    v16bf pa = load_ab(pw + ln * 32 + g * 8, pw + ln * 32 + 16 + g * 8);

    // ---- O += P (16x32) * V (32x64) via 4 WMMAs (V pre-transposed per head)
#pragma unroll
    for (int t = 0; t < 4; ++t) {
      const bf16* vp = Vt + ((size_t)bh * 64 + t * 16 + ln) * (size_t)S + kc + g * 8;
      v16bf vb = load_ab(vp, vp + 16);
      acc[t] = wmma_bf16(pa, vb, acc[t]);
    }
  }

  // ---- normalize and store context in [B,S,D] bf16 (merged heads)
#pragma unroll
  for (int r = 0; r < 8; ++r) {
    float inv = 1.0f / Lr[r];
    size_t base = ((size_t)b * S + q0 + r + 8 * g) * DMODEL + h * 64 + ln;
    ctx[base]      = (bf16)(acc[0][r] * inv);
    ctx[base + 16] = (bf16)(acc[1][r] * inv);
    ctx[base + 32] = (bf16)(acc[2][r] * inv);
    ctx[base + 48] = (bf16)(acc[3][r] * inv);
  }
}

// ---------------------------------------------------------------- launcher
extern "C" void kernel_launch(void* const* d_in, const int* in_sizes, int n_in,
                              void* d_out, int out_size, void* d_ws, size_t ws_size,
                              hipStream_t stream) {
  const float* X   = (const float*)d_in[0];
  const float* Wq  = (const float*)d_in[1];
  const float* Wk  = (const float*)d_in[2];
  const float* Wv  = (const float*)d_in[3];
  const float* Wo  = (const float*)d_in[4];
  const float* rel = (const float*)d_in[5];
  float* out = (float*)d_out;

  const int S = SEQ, D = DMODEL;
  const size_t nX = (size_t)BATCH * S * D;   // 8388608
  const size_t nW = (size_t)D * D;           // 1048576

  char* p = (char*)d_ws;
  bf16* Xb   = (bf16*)p; p += nX * 2;        // reused as ctx after projections
  bf16* Wqb  = (bf16*)p; p += nW * 2;
  bf16* Wkb  = (bf16*)p; p += nW * 2;
  bf16* Wvb  = (bf16*)p; p += nW * 2;
  bf16* Wob  = (bf16*)p; p += nW * 2;
  bf16* Qb   = (bf16*)p; p += nX * 2;
  bf16* Kb   = (bf16*)p; p += nX * 2;
  bf16* Vt   = (bf16*)p; p += nX * 2;
  float* table = (float*)p; p += NHEADS * 4096 * 4;
  bf16* ctx = Xb;

  // 1) casts to bf16
  cast_f32_bf16<<<4096, 256, 0, stream>>>(X,  Xb,  (int)nX);
  cast_f32_bf16<<<1024, 256, 0, stream>>>(Wq, Wqb, (int)nW);
  cast_f32_bf16<<<1024, 256, 0, stream>>>(Wk, Wkb, (int)nW);
  cast_f32_bf16<<<1024, 256, 0, stream>>>(Wv, Wvb, (int)nW);
  cast_f32_bf16<<<1024, 256, 0, stream>>>(Wo, Wob, (int)nW);

  // 2) Q/K/V projections: (8192/128)*(1024/64) = 1024 blocks, 8 waves each
  gemm_bf16_wmma<<<1024, 256, 0, stream>>>(Xb, Wqb, Qb, BATCH * S, D, D, S, 0);
  gemm_bf16_wmma<<<1024, 256, 0, stream>>>(Xb, Wkb, Kb, BATCH * S, D, D, S, 0);
  gemm_bf16_wmma<<<1024, 256, 0, stream>>>(Xb, Wvb, Vt, BATCH * S, D, D, S, 2);

  // 3) relative-position bias table + stream bias output
  build_bias_table<<<256, 256, 0, stream>>>(rel, table);
  write_bias<<<65536, 256, 0, stream>>>(table, out + nX);

  // 4) fused flash attention with bias -> ctx (bf16, merged heads)
  attn_flash<<<dim3(16, BATCH * NHEADS), 256, 0, stream>>>(Qb, Kb, Vt, table, ctx, S);

  // 5) output projection -> fp32 attn_output
  gemm_bf16_wmma<<<1024, 256, 0, stream>>>(ctx, Wob, out, BATCH * S, D, D, S, 1);
}
